// non_local_bb_57260503990844
// MI455X (gfx1250) — matrix operand order repaired and checked by reading, not compiled, not run on hardware
//
#include <hip/hip_runtime.h>

// ---------------------------------------------------------------------------
// Non-local block for MI455X (gfx1250, wave32, WMMA).
//
// N = 10368 voxels, ch = 64, c = 128.
// Stage 1: fused coord-concat + three 1x1x1 conv projections -> f16 tensors
//          fi [N][64], fj [N][64] (voxel-major), fk [64][N] (channel-major),
//          plus Wr converted to f16.
// Stage 2: fused N x N attention, b-dimension split into 6 segments for
//          occupancy (3888 waves). Per wave: 16-row 'a' tile, stream b in
//          chunks of 32: S = (fi@fj)/N via v_wmma_f32_16x16x32_f16,
//          D->A relayout of S through LDS, agg += S@fk. f32 partial agg
//          written to workspace (deterministic, no FP atomics).
// Stage 3: 6-way partial reduction folded into B-operand build, then
//          out = features + Wr @ agg^T + br on WMMA.
// ---------------------------------------------------------------------------

#define NVOX 10368
#define CH 64
#define CFEAT 128
#define CIN 131      // 3 coords + 128 features
#define SEG 6        // b-dimension segments (324 chunks = 6 * 54)
#define BSEG (NVOX / SEG)   // 1728 b per segment, 54 chunks of 32

typedef _Float16 v16h __attribute__((ext_vector_type(16)));
typedef _Float16 v8h  __attribute__((ext_vector_type(8)));
typedef float    v8f  __attribute__((ext_vector_type(8)));

// Concatenate two 16-byte (8 x f16) loads into one 16-element WMMA operand.
static __device__ __forceinline__ v16h ld_cat16(const _Float16* p0, const _Float16* p1) {
  v8h lo = *reinterpret_cast<const v8h*>(p0);
  v8h hi = *reinterpret_cast<const v8h*>(p1);
  return __builtin_shufflevector(lo, hi, 0, 1, 2, 3, 4, 5, 6, 7,
                                          8, 9, 10, 11, 12, 13, 14, 15);
}

static __device__ __forceinline__ v8f wmma_f16(v16h a, v16h b, v8f c) {
  // (neg_a, A, neg_b, B, c_mod, C, reuse_a, reuse_b)
  return __builtin_amdgcn_wmma_f32_16x16x32_f16(false, a, false, b,
                                                (short)0, c, false, false);
}

// ---------------------------------------------------------------------------
// Stage 1: projections. Blocks 0..161 each handle 64 voxels; block 162
// converts Wr to f16.
// ---------------------------------------------------------------------------
__global__ __launch_bounds__(256) void nlb_stage1_proj(
    const float* __restrict__ feat,
    const float* __restrict__ Wi, const float* __restrict__ bi,
    const float* __restrict__ Wj, const float* __restrict__ bj,
    const float* __restrict__ Wk, const float* __restrict__ bk,
    const float* __restrict__ Wr,
    _Float16* __restrict__ fih, _Float16* __restrict__ fjh,
    _Float16* __restrict__ fkc, _Float16* __restrict__ Wrh) {
  if (blockIdx.x == 162) {
    for (int i = threadIdx.x; i < CFEAT * CH; i += 256)
      Wrh[i] = (_Float16)Wr[i];
    return;
  }

  __shared__ float sfc[CIN][64];  // fc tile: [channel][local voxel]
  const int nbase = blockIdx.x * 64;

  // Cooperative, coalesced feature load into LDS.
  for (int i = threadIdx.x; i < CFEAT * 64; i += 256) {
    int c = i >> 6, nl = i & 63;
    sfc[3 + c][nl] = feat[(size_t)c * NVOX + nbase + nl];
  }
  // Coordinate channels (z, y, x each normalized to [0,1] then -0.5).
  if (threadIdx.x < 64) {
    int n = nbase + threadIdx.x;
    int z = n / (24 * 24);
    int rem = n % (24 * 24);
    int y = rem / 24;
    int x = rem % 24;
    sfc[0][threadIdx.x] = (float)z * (1.0f / 17.0f) - 0.5f;
    sfc[1][threadIdx.x] = (float)y * (1.0f / 23.0f) - 0.5f;
    sfc[2][threadIdx.x] = (float)x * (1.0f / 23.0f) - 0.5f;
  }
  __syncthreads();

  // Each thread: one local voxel, 16 output channels, all three projections.
  const int nl = threadIdx.x & 63;
  const int o0 = (threadIdx.x >> 6) * 16;

  float ai[16], aj[16], ak[16];
#pragma unroll
  for (int oo = 0; oo < 16; ++oo) {
    ai[oo] = bi[o0 + oo];
    aj[oo] = bj[o0 + oo];
    ak[oo] = bk[o0 + oo];
  }
  for (int c = 0; c < CIN; ++c) {
    float f = sfc[c][nl];
#pragma unroll
    for (int oo = 0; oo < 16; ++oo) {
      ai[oo] = fmaf(Wi[(o0 + oo) * CIN + c], f, ai[oo]);
      aj[oo] = fmaf(Wj[(o0 + oo) * CIN + c], f, aj[oo]);
      ak[oo] = fmaf(Wk[(o0 + oo) * CIN + c], f, ak[oo]);
    }
  }
  const int n = nbase + nl;
#pragma unroll
  for (int oo = 0; oo < 16; ++oo) {
    int o = o0 + oo;
    fih[(size_t)n * CH + o] = (_Float16)ai[oo];
    fjh[(size_t)n * CH + o] = (_Float16)aj[oo];
    fkc[(size_t)o * NVOX + n] = (_Float16)ak[oo];
  }
}

// ---------------------------------------------------------------------------
// Stage 2: fused attention. One wave per (16-row 'a' tile, b segment):
// 648 * 6 = 3888 waves -> 486 blocks x 8 waves. agg[16][64] in 4 x v8f.
// ---------------------------------------------------------------------------
__global__ __launch_bounds__(256) void nlb_stage2_attn(
    const _Float16* __restrict__ fih, const _Float16* __restrict__ fjh,
    const _Float16* __restrict__ fkc, float* __restrict__ partial) {
  __shared__ __align__(16) _Float16 smem[8][16][32];  // per-wave S staging

  const int lane = threadIdx.x & 31;
  const int w    = threadIdx.x >> 5;
  const int ls   = lane & 15;   // lane within half-wave
  const int hi   = lane >> 4;   // half-wave select
  const int tile = blockIdx.x * 8 + w;
  const int a0   = (tile / SEG) * 16;
  const int seg  = tile % SEG;
  const int bstart = seg * BSEG;
  const float invN = 1.0f / (float)NVOX;

  // A operand (fi rows a0..a0+15), both K-chunks, resident all kernel.
  const _Float16* pi = fih + (size_t)(a0 + ls) * CH;
  v16h aLo = ld_cat16(pi + 16 * hi,      pi + 16 * hi + 8);   // o 0..31
  v16h aHi = ld_cat16(pi + 32 + 16 * hi, pi + 40 + 16 * hi);   // o 32..63

  v8f agg[4];
#pragma unroll
  for (int cc = 0; cc < 4; ++cc) agg[cc] = {};

  // Streaming pointers (advance per 32-b chunk).
  const _Float16* pj  = fjh + (size_t)(bstart + ls) * CH + 16 * hi;
  const _Float16* pk0 = fkc + (size_t)(0  + ls) * NVOX + bstart + 16 * hi;
  const _Float16* pk1 = fkc + (size_t)(16 + ls) * NVOX + bstart + 16 * hi;
  const _Float16* pk2 = fkc + (size_t)(32 + ls) * NVOX + bstart + 16 * hi;
  const _Float16* pk3 = fkc + (size_t)(48 + ls) * NVOX + bstart + 16 * hi;

  for (int it = 0; it < BSEG / 32; ++it) {
    // Issue ALL B-operand tile loads for this chunk up front (distinct
    // registers) so the scheduler can batch them before the first wait.
    v16h bj00 = ld_cat16(pj, pj + 8);                       // cols b0..15,  o 0..31
    v16h bj01 = ld_cat16(pj + 32, pj + 40);                 // cols b0..15,  o 32..63
    v16h bj10 = ld_cat16(pj + 16 * CH, pj + 16 * CH + 8);   // cols b16..31, o 0..31
    v16h bj11 = ld_cat16(pj + 16 * CH + 32, pj + 16 * CH + 40);
    v16h bk0 = ld_cat16(pk0, pk0 + 8);
    v16h bk1 = ld_cat16(pk1, pk1 + 8);
    v16h bk2 = ld_cat16(pk2, pk2 + 8);
    v16h bk3 = ld_cat16(pk3, pk3 + 8);

    __builtin_prefetch(pj + 32 * CH, 0, 3);
    __builtin_prefetch(pk0 + 32, 0, 3);

    // S = fi_tile @ fj_tile  (16a x 32b), f32 accumulation.
    v8f S0 = {}; S0 = wmma_f16(aLo, bj00, S0); S0 = wmma_f16(aHi, bj01, S0);
    v8f S1 = {}; S1 = wmma_f16(aLo, bj10, S1); S1 = wmma_f16(aHi, bj11, S1);

    // Scale by 1/N, narrow to f16, re-layout D -> A through LDS.
#pragma unroll
    for (int r = 0; r < 8; ++r) {
      smem[w][r + 8 * hi][ls]      = (_Float16)(S0[r] * invN);
      smem[w][r + 8 * hi][16 + ls] = (_Float16)(S1[r] * invN);
    }
    asm volatile("s_wait_dscnt 0" ::: "memory");
    v16h a2 = ld_cat16(&smem[w][ls][16 * hi], &smem[w][ls][16 * hi + 8]);

    // agg(16a x 64c) += S(16a x 32b) @ fk(32b x 64c).
    agg[0] = wmma_f16(a2, bk0, agg[0]);
    agg[1] = wmma_f16(a2, bk1, agg[1]);
    agg[2] = wmma_f16(a2, bk2, agg[2]);
    agg[3] = wmma_f16(a2, bk3, agg[3]);

    pj  += 32 * CH;
    pk0 += 32; pk1 += 32; pk2 += 32; pk3 += 32;
  }

  // Deterministic f32 partial store: partial[seg][N][64].
  float* pp = partial + (size_t)seg * NVOX * CH;
#pragma unroll
  for (int cc = 0; cc < 4; ++cc) {
#pragma unroll
    for (int r = 0; r < 8; ++r) {
      pp[(size_t)(a0 + r + 8 * hi) * CH + cc * 16 + ls] = agg[cc][r];
    }
  }
}

// ---------------------------------------------------------------------------
// Stage 3: reduce 6 partials into the B operand (f32 sum -> f16), then
// out[128][N] = features + Wr @ agg^T + br. One wave per 16-column n tile.
// ---------------------------------------------------------------------------
__global__ __launch_bounds__(256) void nlb_stage3_out(
    const float* __restrict__ feat, const _Float16* __restrict__ Wrh,
    const float* __restrict__ br, const float* __restrict__ partial,
    float* __restrict__ out) {
  const int lane = threadIdx.x & 31;
  const int w    = threadIdx.x >> 5;
  const int ls   = lane & 15;
  const int hi   = lane >> 4;
  const int n0   = (blockIdx.x * 8 + w) * 16;

  // Build B operand = agg^T tiles (K = c): sum SEG partials, convert to f16.
  const float* pb = partial + (size_t)(n0 + ls) * CH + 16 * hi;
  v16h bA0, bA1;
#pragma unroll
  for (int e = 0; e < 16; ++e) {
    float s0 = 0.0f, s1 = 0.0f;
#pragma unroll
    for (int s = 0; s < SEG; ++s) {
      const float* ps = pb + (size_t)s * NVOX * CH;
      s0 += ps[e];
      s1 += ps[32 + e];
    }
    bA0[e] = (_Float16)s0;   // c 0..31 chunk
    bA1[e] = (_Float16)s1;   // c 32..63 chunk
  }

#pragma unroll
  for (int cot = 0; cot < 8; ++cot) {
    const int co0 = cot * 16;
    const _Float16* pa = Wrh + (size_t)(co0 + ls) * CH;
    v16h A0 = ld_cat16(pa + 16 * hi,      pa + 16 * hi + 8);
    v16h A1 = ld_cat16(pa + 32 + 16 * hi, pa + 40 + 16 * hi);

    v8f acc = {};
    acc = wmma_f16(A0, bA0, acc);
    acc = wmma_f16(A1, bA1, acc);

#pragma unroll
    for (int r = 0; r < 8; ++r) {
      int co = co0 + r + 8 * hi;
      size_t idx = (size_t)co * NVOX + n0 + ls;
      out[idx] = feat[idx] + br[co] + acc[r];
    }
  }
}

// ---------------------------------------------------------------------------
// Launcher. Workspace: fi | fj | fk (f16, 1.33 MB each) | Wr_h (16 KB) |
// partial agg f32 [SEG][N][64] (15.9 MB). Total ~20 MB, L2-resident.
// ---------------------------------------------------------------------------
extern "C" void kernel_launch(void* const* d_in, const int* in_sizes, int n_in,
                              void* d_out, int out_size, void* d_ws, size_t ws_size,
                              hipStream_t stream) {
  const float* feat = (const float*)d_in[0];
  const float* Wi   = (const float*)d_in[1];
  const float* bi   = (const float*)d_in[2];
  const float* Wj   = (const float*)d_in[3];
  const float* bj   = (const float*)d_in[4];
  const float* Wk   = (const float*)d_in[5];
  const float* bk   = (const float*)d_in[6];
  const float* Wr   = (const float*)d_in[7];
  const float* br   = (const float*)d_in[8];
  float* out = (float*)d_out;

  char* ws = (char*)d_ws;
  const size_t projBytes = (size_t)NVOX * CH * sizeof(_Float16);
  _Float16* fih  = (_Float16*)(ws);
  _Float16* fjh  = (_Float16*)(ws + projBytes);
  _Float16* fkc  = (_Float16*)(ws + 2 * projBytes);
  _Float16* Wrh  = (_Float16*)(ws + 3 * projBytes);
  float*    part = (float*)(ws + 3 * projBytes + 16384);

  // 162 voxel-tile blocks + 1 block for Wr f16 conversion.
  nlb_stage1_proj<<<163, 256, 0, stream>>>(feat, Wi, bi, Wj, bj, Wk, bk, Wr,
                                           fih, fjh, fkc, Wrh);
  // 648 a-tiles * 6 b-segments = 3888 waves / 8 per block.
  nlb_stage2_attn<<<486, 256, 0, stream>>>(fih, fjh, fkc, part);
  // 648 n-tiles / 8 waves per block.
  nlb_stage3_out<<<81, 256, 0, stream>>>(feat, Wrh, br, part, out);
}
